// Attention_5772436046577
// MI455X (gfx1250) — matrix operand order repaired and checked by reading, not compiled, not run on hardware
//
#include <hip/hip_runtime.h>
#include <math.h>

// ---------------------------------------------------------------------------
// Fixed problem geometry (from setup_inputs) -- constexpr so all row strides
// fold to shifts/immediates (no v_mul_u64 in hot loops).
// ---------------------------------------------------------------------------
constexpr int T_ = 2048, H_ = 16, KH_ = 2, D_ = 128;
constexpr int G_ = H_ / KH_;
constexpr int KSZ_ = 32, STRIDE_ = 16, BSZ_ = 64;
constexpr int TOPN_ = 16, NINIT_ = 1, NLOCAL_ = 2, WINDOW_ = 512;
constexpr int NCMP_ = (T_ - KSZ_) / STRIDE_ + 1;  // 127
constexpr int NCP_ = 128;                         // padded compressed keys
constexpr int NBLK_ = (T_ + BSZ_ - 1) / BSZ_;     // 32

typedef __attribute__((ext_vector_type(16))) __bf16 v16bf;
typedef __attribute__((ext_vector_type(8)))  __bf16 v8bf;
typedef __attribute__((ext_vector_type(8)))  float  v8f;

__device__ __forceinline__ __bf16 to_bf16(float f) {
  unsigned u = __float_as_uint(f);
  u += 0x7FFFu + ((u >> 16) & 1u);  // round-to-nearest-even
  return __builtin_bit_cast(__bf16, (unsigned short)(u >> 16));
}

__device__ __forceinline__ v8bf ld8(const __bf16* p) {
  return *(const v8bf*)p;  // 16-byte aligned by construction
}

__device__ __forceinline__ v16bf cat16(v8bf lo, v8bf hi) {
  return __builtin_shufflevector(lo, hi, 0, 1, 2, 3, 4, 5, 6, 7, 8, 9, 10, 11,
                                 12, 13, 14, 15);
}

__device__ __forceinline__ v8f wmma_bf16(v16bf a, v16bf b, v8f c) {
  // D = A(16x32 bf16) * B(32x16 bf16) + C(16x16 f32)
  return __builtin_amdgcn_wmma_f32_16x16x32_bf16(false, a, false, b, (short)0, c,
                                                 false, false);
}

// A-fragment (16-bit A 16x32): element e -> K = half*16+e (contiguous 16)
__device__ __forceinline__ v16bf load_a_frag(const __bf16* row, int half) {
  return cat16(ld8(row + (half << 4)), ld8(row + 8 + (half << 4)));
}
// B-fragment (16-bit B 32x16): element e -> K = half*16+e (contiguous 16)
__device__ __forceinline__ v16bf load_b_frag(const __bf16* p) {
  return cat16(ld8(p), ld8(p + 8));
}

// ---------------------------------------------------------------------------
// Kernel 0a: q (T,H,D) f32 -> qbf (T,H,D) bf16.  Grid-stride, 4 elems/thread.
// ---------------------------------------------------------------------------
__global__ __launch_bounds__(256) void cvt_q_kernel(const float* __restrict__ q,
                                                    __bf16* __restrict__ qbf) {
  constexpr int n4 = T_ * H_ * D_ / 4;
  for (int i = blockIdx.x * blockDim.x + threadIdx.x; i < n4;
       i += gridDim.x * blockDim.x) {
    const float4 f = ((const float4*)q)[i];
    __bf16* o = qbf + (size_t)i * 4;
    o[0] = to_bf16(f.x); o[1] = to_bf16(f.y);
    o[2] = to_bf16(f.z); o[3] = to_bf16(f.w);
  }
}

// ---------------------------------------------------------------------------
// Kernel 0b: k (T,KH,D) -> kbf[kh][t][d] bf16 ; v -> vT[kh][d][t] bf16.
// grid(T, KH), block(D)
// ---------------------------------------------------------------------------
__global__ __launch_bounds__(128) void cvt_kv_kernel(
    const float* __restrict__ k, const float* __restrict__ v,
    __bf16* __restrict__ kbf, __bf16* __restrict__ vT) {
  int t = blockIdx.x, kh = blockIdx.y, d = threadIdx.x;
  size_t src = (size_t)t * KH_ * D_ + (size_t)kh * D_ + d;
  kbf[(size_t)kh * T_ * D_ + (size_t)t * D_ + d] = to_bf16(k[src]);
  vT[(size_t)kh * D_ * T_ + (size_t)d * T_ + t] = to_bf16(v[src]);
}

// ---------------------------------------------------------------------------
// Kernel 1: mean-pool compressed K/V -> bf16, padded to NCP_ rows.
// cmpk[kh][j][d], cmpvT[kh][d][j].  grid(NCP_, KH), block(D)
// ---------------------------------------------------------------------------
__global__ __launch_bounds__(128) void cmpkv_kernel(
    const float* __restrict__ k, const float* __restrict__ v,
    __bf16* __restrict__ cmpk, __bf16* __restrict__ cmpvT) {
  int j = blockIdx.x, kh = blockIdx.y, d = threadIdx.x;
  float sk = 0.f, sv = 0.f;
  if (j < NCMP_) {
    int base = j * STRIDE_;
    for (int i = 0; i < KSZ_; ++i) {
      size_t off = (size_t)(base + i) * KH_ * D_ + (size_t)kh * D_ + d;
      sk += k[off];
      sv += v[off];
    }
    constexpr float inv = 1.0f / (float)KSZ_;
    sk *= inv; sv *= inv;
  }
  cmpk[(size_t)kh * NCP_ * D_ + (size_t)j * D_ + d] = to_bf16(sk);
  cmpvT[(size_t)kh * D_ * NCP_ + (size_t)d * NCP_ + j] = to_bf16(sv);
}

// ---------------------------------------------------------------------------
// Kernel 2: compressed attention (WMMA).  grid(T/16, H), block(32) = 1 wave.
// Writes out = w0 * o_cmp ; atomically accumulates blk_score over g-heads.
// ---------------------------------------------------------------------------
__global__ __launch_bounds__(32) void cmp_attn_kernel(
    const __bf16* __restrict__ qbf, const __bf16* __restrict__ cmpk,
    const __bf16* __restrict__ cmpvT, const float* __restrict__ w,
    float* __restrict__ out, float* __restrict__ blk_score) {
  const int t0 = blockIdx.x * 16;
  const int h = blockIdx.y;
  const int kh = h / G_;
  const int lane = threadIdx.x;
  const int nlo = lane & 15, half = lane >> 4;
  const float scale = 1.0f / sqrtf((float)D_);

  __shared__ float  lsc[16][32];
  __shared__ __bf16 lds_p[16][32];

  const __bf16* kcb = cmpk + (size_t)kh * NCP_ * D_;
  const __bf16* vcb = cmpvT + (size_t)kh * D_ * NCP_;
  const __bf16* qrow = qbf + (size_t)(t0 + nlo) * H_ * D_ + (size_t)h * D_;

  v16bf qa[4];
#pragma unroll
  for (int dc = 0; dc < 4; ++dc) qa[dc] = load_a_frag(qrow + dc * 32, half);

  // Logits for 4 chunks x 2 subtiles (covers NCP_=128 compressed keys)
  v8f S[4][2];
#pragma unroll
  for (int cc = 0; cc < 4; ++cc)
#pragma unroll
    for (int tn = 0; tn < 2; ++tn)
#pragma unroll
      for (int e = 0; e < 8; ++e) S[cc][tn][e] = 0.f;

#pragma unroll
  for (int cc = 0; cc < 4; ++cc) {
#pragma unroll
    for (int dc = 0; dc < 4; ++dc) {
#pragma unroll
      for (int tn = 0; tn < 2; ++tn) {
        int key = (cc << 5) + (tn << 4) + nlo;  // < NCP_ (padded)
        v16bf bk = load_b_frag(kcb + (size_t)key * D_ + dc * 32 + (half << 4));
        S[cc][tn] = wmma_bf16(qa[dc], bk, S[cc][tn]);
      }
    }
  }

  // Mask + exact softmax (two pass, registers)
  float mrow[8];
#pragma unroll
  for (int r = 0; r < 8; ++r) mrow[r] = -3.0e38f;
#pragma unroll
  for (int cc = 0; cc < 4; ++cc)
#pragma unroll
    for (int tn = 0; tn < 2; ++tn)
#pragma unroll
      for (int r = 0; r < 8; ++r) {
        int j = (cc << 5) + (tn << 4) + nlo;
        int t = t0 + r + (half << 3);
        float sv = S[cc][tn][r] * scale;
        bool ok = (j < NCMP_) && (j * STRIDE_ + KSZ_ - 1 <= t);
        sv = ok ? sv : -3.0e38f;
        S[cc][tn][r] = sv;
        mrow[r] = fmaxf(mrow[r], sv);
      }
#pragma unroll
  for (int r = 0; r < 8; ++r)
#pragma unroll
    for (int off = 1; off < 16; off <<= 1)
      mrow[r] = fmaxf(mrow[r], __shfl_xor(mrow[r], off, 32));

  float lrow[8];
#pragma unroll
  for (int r = 0; r < 8; ++r) lrow[r] = 0.f;
#pragma unroll
  for (int cc = 0; cc < 4; ++cc)
#pragma unroll
    for (int tn = 0; tn < 2; ++tn)
#pragma unroll
      for (int r = 0; r < 8; ++r) {
        float p = (S[cc][tn][r] > -1.0e30f) ? __expf(S[cc][tn][r] - mrow[r]) : 0.f;
        S[cc][tn][r] = p;
        lrow[r] += p;
      }
#pragma unroll
  for (int r = 0; r < 8; ++r) {
#pragma unroll
    for (int off = 1; off < 16; off <<= 1) lrow[r] += __shfl_xor(lrow[r], off, 32);
    float inv = (lrow[r] > 0.f) ? (1.0f / lrow[r]) : 0.f;
#pragma unroll
    for (int cc = 0; cc < 4; ++cc)
#pragma unroll
      for (int tn = 0; tn < 2; ++tn) S[cc][tn][r] *= inv;
  }

  // Block scores: LDS accumulate then one global atomicAdd per (row, block)
  for (int i = lane; i < 512; i += 32) (&lsc[0][0])[i] = 0.f;
  __syncthreads();
#pragma unroll
  for (int cc = 0; cc < 4; ++cc)
#pragma unroll
    for (int tn = 0; tn < 2; ++tn)
#pragma unroll
      for (int r = 0; r < 8; ++r) {
        int j = (cc << 5) + (tn << 4) + nlo;
        float p = S[cc][tn][r];
        if (j < NCMP_ && p > 0.f) {
          int m = r + (half << 3);
          int b0 = (j * STRIDE_) / BSZ_;
          int b1 = (j * STRIDE_ + KSZ_ - 1) / BSZ_;
          for (int b = b0; b <= b1; ++b)
            if (b < NBLK_) atomicAdd(&lsc[m][b], p);
        }
      }
  __syncthreads();
  for (int i = lane; i < 512; i += 32) {
    int m = i >> 5, b = i & 31;
    float val = lsc[m][b];
    if (val != 0.f)
      atomicAdd(&blk_score[(size_t)kh * T_ * NBLK_ + (size_t)(t0 + m) * NBLK_ + b],
                val);
  }

  // o_cmp = P (normalized) @ Vc
  v8f acc[8];
#pragma unroll
  for (int dt = 0; dt < 8; ++dt)
#pragma unroll
    for (int e = 0; e < 8; ++e) acc[dt][e] = 0.f;

#pragma unroll
  for (int cc = 0; cc < 4; ++cc) {
    __syncthreads();
#pragma unroll
    for (int r = 0; r < 8; ++r) {
      lds_p[r + (half << 3)][nlo]      = to_bf16(S[cc][0][r]);
      lds_p[r + (half << 3)][16 + nlo] = to_bf16(S[cc][1][r]);
    }
    __syncthreads();
    v16bf pf = load_a_frag(&lds_p[nlo][0], half);
#pragma unroll
    for (int dt = 0; dt < 8; ++dt) {
      v16bf bv = load_b_frag(vcb + (size_t)(dt * 16 + nlo) * NCP_ + (cc << 5) +
                             (half << 4));
      acc[dt] = wmma_bf16(pf, bv, acc[dt]);
    }
  }

#pragma unroll
  for (int r = 0; r < 8; ++r) {
    int t = t0 + r + (half << 3);
    float w0 = w[(size_t)t * H_ * 3 + (size_t)h * 3 + 0];
#pragma unroll
    for (int dt = 0; dt < 8; ++dt)
      out[(size_t)t * H_ * D_ + (size_t)h * D_ + dt * 16 + nlo] = w0 * acc[dt][r];
  }
}

// ---------------------------------------------------------------------------
// Kernel 3: per-(kh, t) top-k block selection -> 32-bit mask.
// sc[] is read-only after init -> stays in registers; outer loop stays rolled.
// ---------------------------------------------------------------------------
__global__ __launch_bounds__(256) void topk_kernel(
    const float* __restrict__ blk_score, unsigned* __restrict__ sel) {
  int idx = blockIdx.x * blockDim.x + threadIdx.x;
  if (idx >= KH_ * T_) return;
  int kh = idx / T_, t = idx - kh * T_;
  int t_blk = t / BSZ_;
  float sc[NBLK_];
#pragma unroll
  for (int b = 0; b < NBLK_; ++b) {
    float s = blk_score[(size_t)kh * T_ * NBLK_ + (size_t)t * NBLK_ + b];
    bool forced = (b < NINIT_) || ((b <= t_blk) && (b >= t_blk - NLOCAL_ + 1));
    bool valid = b * BSZ_ <= t;
    s = forced ? 1.0e30f : s;
    s = valid ? s : -1.0e30f;
    sc[b] = s;
  }
  unsigned mask = 0;
#pragma unroll 1
  for (int i = 0; i < TOPN_; ++i) {
    int bj = 0;
    float best = -3.9e38f;
#pragma unroll
    for (int b = 0; b < NBLK_; ++b) {
      bool take = !((mask >> b) & 1u) && (sc[b] > best);  // ties -> lowest (jax)
      best = take ? sc[b] : best;
      bj = take ? b : bj;
    }
    mask |= (1u << bj);
  }
  sel[(size_t)kh * T_ + t] = mask;
}

// ---------------------------------------------------------------------------
// Kernels 4/5: flash attention over full K/V with mask.
// MODE 0: block-selected (sel bitmask & causal), out += w1 * o_slc
// MODE 1: sliding window (causal & t-s<window),  out += w2 * o_win
// grid(T/16, H), block(32) = 1 wave.
// ---------------------------------------------------------------------------
template <int MODE>
__global__ __launch_bounds__(32) void flash_kernel(
    const __bf16* __restrict__ qbf, const __bf16* __restrict__ kbf,
    const __bf16* __restrict__ vT, const float* __restrict__ w,
    const unsigned* __restrict__ sel, float* __restrict__ out) {
  const int t0 = blockIdx.x * 16;
  const int h = blockIdx.y;
  const int kh = h / G_;
  const int lane = threadIdx.x;
  const int nlo = lane & 15, half = lane >> 4;
  const float scale = 1.0f / sqrtf((float)D_);

  __shared__ __bf16 lds_p[16][32];

  const __bf16* qrow = qbf + (size_t)(t0 + nlo) * H_ * D_ + (size_t)h * D_;
  v16bf qa[4];
#pragma unroll
  for (int dc = 0; dc < 4; ++dc) qa[dc] = load_a_frag(qrow + dc * 32, half);

  v8f acc[8];
#pragma unroll
  for (int dt = 0; dt < 8; ++dt)
#pragma unroll
    for (int e = 0; e < 8; ++e) acc[dt][e] = 0.f;

  float mrow[8], lrow[8];
#pragma unroll
  for (int r = 0; r < 8; ++r) { mrow[r] = -3.0e38f; lrow[r] = 0.f; }

  unsigned selrow[8] = {};
  unsigned usel = 0;
  if (MODE == 0) {
#pragma unroll
    for (int r = 0; r < 8; ++r)
      selrow[r] = sel[(size_t)kh * T_ + (t0 + r + (half << 3))];
    unsigned u = 0;
#pragma unroll
    for (int r = 0; r < 8; ++r) u |= selrow[r];
    u |= __shfl_xor(u, 16, 32);  // union across both row-halves -> uniform
    usel = u;
  }

  const __bf16* kbase = kbf + (size_t)kh * T_ * D_;
  const __bf16* vbase = vT + (size_t)kh * D_ * T_;
  const int nchunks = ((t0 + 15) >> 5) + 1;

  for (int c = 0; c < nchunks; ++c) {
    const int kb = c << 5;
    if (MODE == 0) {
      if (!((usel >> (kb / BSZ_)) & 1u)) continue;  // no row selects block
    } else {
      if (kb + 31 < t0 - (WINDOW_ - 1)) continue;   // fully out of window
    }

    // Prefetch next chunk's K rows (global_prefetch; no LOADcnt impact)
    if (kb + 63 < T_) {
      __builtin_prefetch(kbase + (size_t)(kb + 32 + nlo) * D_, 0, 1);
      __builtin_prefetch(kbase + (size_t)(kb + 48 + nlo) * D_, 0, 1);
    }

    // S = Q K^T for 32 keys (two 16-col subtiles)
    v8f S0, S1;
#pragma unroll
    for (int e = 0; e < 8; ++e) { S0[e] = 0.f; S1[e] = 0.f; }
#pragma unroll
    for (int dc = 0; dc < 4; ++dc) {
      v16bf b0 =
          load_b_frag(kbase + (size_t)(kb + nlo) * D_ + dc * 32 + (half << 4));
      v16bf b1 =
          load_b_frag(kbase + (size_t)(kb + 16 + nlo) * D_ + dc * 32 + (half << 4));
      S0 = wmma_bf16(qa[dc], b0, S0);
      S1 = wmma_bf16(qa[dc], b1, S1);
    }

    // Mask + scale
    float cmax[8];
#pragma unroll
    for (int r = 0; r < 8; ++r) {
      const int t = t0 + r + (half << 3);
      {
        int s = kb + nlo;
        float sv = S0[r] * scale;
        bool ok = (s <= t);
        if (MODE == 0) ok = ok && (((selrow[r] >> (s / BSZ_)) & 1u) != 0u);
        else           ok = ok && ((t - s) < WINDOW_);
        sv = ok ? sv : -3.0e38f;
        S0[r] = sv;
        cmax[r] = sv;
      }
      {
        int s = kb + 16 + nlo;
        float sv = S1[r] * scale;
        bool ok = (s <= t);
        if (MODE == 0) ok = ok && (((selrow[r] >> (s / BSZ_)) & 1u) != 0u);
        else           ok = ok && ((t - s) < WINDOW_);
        sv = ok ? sv : -3.0e38f;
        S1[r] = sv;
        cmax[r] = fmaxf(cmax[r], sv);
      }
    }
#pragma unroll
    for (int r = 0; r < 8; ++r)
#pragma unroll
      for (int off = 1; off < 16; off <<= 1)
        cmax[r] = fmaxf(cmax[r], __shfl_xor(cmax[r], off, 32));

    // Online softmax update; unnormalized P -> LDS (bf16)
    float alpha[8], psum[8];
#pragma unroll
    for (int r = 0; r < 8; ++r) {
      float mnew = fmaxf(mrow[r], cmax[r]);
      alpha[r] = __expf(mrow[r] - mnew);  // -3e38 - finite -> underflow -> 0
      mrow[r] = mnew;
      float p0 = (S0[r] > -1.0e30f) ? __expf(S0[r] - mnew) : 0.f;
      float p1 = (S1[r] > -1.0e30f) ? __expf(S1[r] - mnew) : 0.f;
      psum[r] = p0 + p1;
      lds_p[r + (half << 3)][nlo]      = to_bf16(p0);
      lds_p[r + (half << 3)][16 + nlo] = to_bf16(p1);
    }
#pragma unroll
    for (int r = 0; r < 8; ++r) {
#pragma unroll
      for (int off = 1; off < 16; off <<= 1) psum[r] += __shfl_xor(psum[r], off, 32);
      lrow[r] = lrow[r] * alpha[r] + psum[r];
#pragma unroll
      for (int dt = 0; dt < 8; ++dt) acc[dt][r] *= alpha[r];
    }
    __syncthreads();

    v16bf pf = load_a_frag(&lds_p[nlo][0], half);
#pragma unroll
    for (int dt = 0; dt < 8; ++dt) {
      v16bf bv =
          load_b_frag(vbase + (size_t)(dt * 16 + nlo) * T_ + kb + (half << 4));
      acc[dt] = wmma_bf16(pf, bv, acc[dt]);
    }
    __syncthreads();
  }

  // Epilogue: out += w[mode] * o / l
  const int widx = (MODE == 0) ? 1 : 2;
#pragma unroll
  for (int r = 0; r < 8; ++r) {
    const int t = t0 + r + (half << 3);
    const float inv = (lrow[r] > 0.f) ? (1.0f / lrow[r]) : 0.f;
    const float wgt = w[(size_t)t * H_ * 3 + (size_t)h * 3 + widx] * inv;
#pragma unroll
    for (int dt = 0; dt < 8; ++dt) {
      size_t o = (size_t)t * H_ * D_ + (size_t)h * D_ + dt * 16 + nlo;
      out[o] += wgt * acc[dt][r];
    }
  }
}

// ---------------------------------------------------------------------------
extern "C" void kernel_launch(void* const* d_in, const int* in_sizes, int n_in,
                              void* d_out, int out_size, void* d_ws, size_t ws_size,
                              hipStream_t stream) {
  (void)in_sizes; (void)n_in; (void)out_size; (void)ws_size;
  const float* q = (const float*)d_in[0];
  const float* k = (const float*)d_in[1];
  const float* v = (const float*)d_in[2];
  const float* w = (const float*)d_in[3];

  // Workspace layout (bf16 staging + scores + selection masks)
  char* p = (char*)d_ws;
  __bf16* qbf = (__bf16*)p;   p += (size_t)T_ * H_ * D_ * 2;
  __bf16* kbf = (__bf16*)p;   p += (size_t)KH_ * T_ * D_ * 2;
  __bf16* vTbf = (__bf16*)p;  p += (size_t)KH_ * D_ * T_ * 2;
  __bf16* cmpk = (__bf16*)p;  p += (size_t)KH_ * NCP_ * D_ * 2;
  __bf16* cmpvT = (__bf16*)p; p += (size_t)KH_ * D_ * NCP_ * 2;
  float* blk = (float*)p;     p += (size_t)KH_ * T_ * NBLK_ * 4;
  unsigned* sel = (unsigned*)p;
  float* out = (float*)d_out;

  constexpr int n4 = T_ * H_ * D_ / 4;
  cvt_q_kernel<<<(n4 + 255) / 256, 256, 0, stream>>>(q, qbf);
  cvt_kv_kernel<<<dim3(T_, KH_), D_, 0, stream>>>(k, v, kbf, vTbf);
  cmpkv_kernel<<<dim3(NCP_, KH_), D_, 0, stream>>>(k, v, cmpk, cmpvT);
  hipMemsetAsync(blk, 0, (size_t)KH_ * T_ * NBLK_ * sizeof(float), stream);
  cmp_attn_kernel<<<dim3(T_ / 16, H_), 32, 0, stream>>>(qbf, cmpk, cmpvT, w, out,
                                                        blk);
  topk_kernel<<<(KH_ * T_ + 255) / 256, 256, 0, stream>>>(blk, sel);
  flash_kernel<0><<<dim3(T_ / 16, H_), 32, 0, stream>>>(qbf, kbf, vTbf, w, sel,
                                                        out);
  flash_kernel<1><<<dim3(T_ / 16, H_), 32, 0, stream>>>(qbf, kbf, vTbf, w, sel,
                                                        out);
}
